// HeteroGNN_27015344292138
// MI455X (gfx1250) — matrix operand order repaired and checked by reading, not compiled, not run on hardware
//
#include <hip/hip_runtime.h>
#include <hip/hip_bf16.h>

typedef __attribute__((ext_vector_type(16))) _Float16 v16h;
typedef __attribute__((ext_vector_type(8)))  float    v8f;

#define NA_N 100000
#define NP_N 200000
#define NU_N 5000
#define HDIM 32
#define EW_N 2000000
#define EP_N 200000
#define NEG_SLOPE 0.2f

// ---------------------------------------------------------------------------
// ordered-uint mapping for float atomic max
// ---------------------------------------------------------------------------
__device__ __forceinline__ unsigned fmap_ord(float f) {
    unsigned u = __float_as_uint(f);
    return (u & 0x80000000u) ? ~u : (u | 0x80000000u);
}
__device__ __forceinline__ float funmap_ord(unsigned u) {
    return __uint_as_float((u & 0x80000000u) ? (u ^ 0x80000000u) : ~u);
}

// ---------------------------------------------------------------------------
// Fused projection + attention score:
//   h = fp32( f16(X[N x D]) @ f16(W[D x 32]) )   (optional store)
//   s = h @ att                                   (in-register shuffle reduce)
// One wave computes a 16-row x 32-col tile via two v_wmma_f32_16x16x32_f16
// per 32-wide K chunk. 8 waves / block -> 128 rows per block.
// ---------------------------------------------------------------------------
__global__ __launch_bounds__(256)
void proj_score_kernel(const float* __restrict__ X, const float* __restrict__ W,
                       const float* __restrict__ att, int N, int D,
                       float* __restrict__ h_out, float* __restrict__ s_out)
{
    const int lane = threadIdx.x & 31;
    const int wave = threadIdx.x >> 5;
    const int baseRow = (blockIdx.x * 8 + wave) * 16;
    if (baseRow >= N) return;

    const int r15  = lane & 15;
    const int half = lane >> 4;

    v8f acc0, acc1;
#pragma unroll
    for (int i = 0; i < 8; ++i) { acc0[i] = 0.0f; acc1[i] = 0.0f; }

    const int  rowA  = baseRow + r15;
    const bool rowOk = rowA < N;
    const float* xrow = X + (size_t)rowA * D;

    for (int kb = 0; kb < D; kb += 32) {
        // --- A fragment: 16x32 f16, lanes 0-15 K=0..15, 16-31 K=16..31
        v16h a;
        if (rowOk) {
            const float4* p0 = (const float4*)(xrow + kb + half * 16);
            const float4* p1 = (const float4*)(xrow + kb + half * 16 + 8);
            float4 f0 = p0[0], f1 = p0[1], f2 = p1[0], f3 = p1[1];
            a[0]=(_Float16)f0.x;  a[1]=(_Float16)f0.y;  a[2]=(_Float16)f0.z;  a[3]=(_Float16)f0.w;
            a[4]=(_Float16)f1.x;  a[5]=(_Float16)f1.y;  a[6]=(_Float16)f1.z;  a[7]=(_Float16)f1.w;
            a[8]=(_Float16)f2.x;  a[9]=(_Float16)f2.y;  a[10]=(_Float16)f2.z; a[11]=(_Float16)f2.w;
            a[12]=(_Float16)f3.x; a[13]=(_Float16)f3.y; a[14]=(_Float16)f3.z; a[15]=(_Float16)f3.w;
        } else {
#pragma unroll
            for (int e = 0; e < 16; ++e) a[e] = (_Float16)0.0f;
        }
        // --- B fragments: 32x16 f16, lane = column, lanes 0-15 K=0..15, 16-31 K=16..31
        v16h b0, b1;
        const float* wbase = W + (size_t)(kb + half * 16) * HDIM + r15;
#pragma unroll
        for (int e = 0; e < 16; ++e) {
            b0[e] = (_Float16)wbase[e * HDIM];
            b1[e] = (_Float16)wbase[e * HDIM + 16];
        }
        acc0 = __builtin_amdgcn_wmma_f32_16x16x32_f16(false, a, false, b0, (short)0, acc0, false, false);
        acc1 = __builtin_amdgcn_wmma_f32_16x16x32_f16(false, a, false, b1, (short)0, acc1, false, false);
    }

    // --- attention score: s[row] = sum_c h[row][c] * att[c]
    const float av0 = att[r15];
    const float av1 = att[16 + r15];
    float part[8];
#pragma unroll
    for (int r = 0; r < 8; ++r) part[r] = acc0[r] * av0 + acc1[r] * av1;
#pragma unroll
    for (int off = 1; off < 16; off <<= 1) {
#pragma unroll
        for (int r = 0; r < 8; ++r) part[r] += __shfl_xor(part[r], off, 32);
    }
#pragma unroll
    for (int r = 0; r < 8; ++r) {
        int row = baseRow + half * 8 + r;
        if (r15 == r && row < N) s_out[row] = part[r];
    }

    // --- optional h store (C/D layout: VGPR r -> row base + r + half*8)
    if (h_out) {
#pragma unroll
        for (int r = 0; r < 8; ++r) {
            int row = baseRow + half * 8 + r;
            if (row < N) {
                h_out[(size_t)row * HDIM + r15]      = acc0[r];
                h_out[(size_t)row * HDIM + 16 + r15] = acc1[r];
            }
        }
    }
}

// ---------------------------------------------------------------------------
// Pass 1: per-edge leaky-relu score, atomic segment-max over dst (ordered uint)
// ---------------------------------------------------------------------------
__global__ void edge_max_kernel(const int* __restrict__ src, const int* __restrict__ dst,
                                const float* __restrict__ ssrc, const float* __restrict__ sdst,
                                unsigned* __restrict__ mmax, int E)
{
    int i = blockIdx.x * blockDim.x + threadIdx.x;
    if (i >= E) return;
    int d = dst[i];
    float e = ssrc[src[i]] + sdst[d];
    e = e > 0.0f ? e : NEG_SLOPE * e;
    atomicMax(&mmax[d], fmap_ord(e));
}

// ---------------------------------------------------------------------------
// Pass 2: ex = exp(e - m[dst]); den[dst] += ex; accum[dst] += ex * h[src]
// 4 threads per edge, 8 columns each.
// ---------------------------------------------------------------------------
__global__ void edge_accum_kernel(const int* __restrict__ src, const int* __restrict__ dst,
                                  const float* __restrict__ ssrc, const float* __restrict__ sdst,
                                  const unsigned* __restrict__ mmax, const float* __restrict__ hsrc,
                                  float* __restrict__ den, float* __restrict__ accum, int E)
{
    int gid = blockIdx.x * blockDim.x + threadIdx.x;
    int i = gid >> 2, q = gid & 3;
    if (i >= E) return;
    int s = src[i], d = dst[i];
    float e = ssrc[s] + sdst[d];
    e = e > 0.0f ? e : NEG_SLOPE * e;
    float ex = __expf(e - funmap_ord(mmax[d]));
    if (q == 0) atomicAdd(&den[d], ex);
    const float4* hp = (const float4*)(hsrc + (size_t)s * HDIM + q * 8);
    float4 h0 = hp[0], h1 = hp[1];
    float* ap = accum + (size_t)d * HDIM + q * 8;
    atomicAdd(ap + 0, h0.x * ex); atomicAdd(ap + 1, h0.y * ex);
    atomicAdd(ap + 2, h0.z * ex); atomicAdd(ap + 3, h0.w * ex);
    atomicAdd(ap + 4, h1.x * ex); atomicAdd(ap + 5, h1.y * ex);
    atomicAdd(ap + 6, h1.z * ex); atomicAdd(ap + 7, h1.w * ex);
}

// ---------------------------------------------------------------------------
// Per-node normalize + bias (in place). den==0 (no in-edges) -> just bias.
// ---------------------------------------------------------------------------
__global__ void finalize_kernel(float* __restrict__ out, const float* __restrict__ den,
                                const float* __restrict__ bias, int N)
{
    int t = blockIdx.x * blockDim.x + threadIdx.x;
    if (t >= N * HDIM) return;
    int node = t >> 5, c = t & 31;
    float dd = den[node];
    float v  = out[t];
    out[t] = (dd > 0.0f ? v / dd : 0.0f) + bias[c];
}

__global__ void finalize_paper_kernel(float* __restrict__ p1, const float* __restrict__ p2,
                                      const float* __restrict__ den1, const float* __restrict__ den2,
                                      const float* __restrict__ b1, const float* __restrict__ b2, int N)
{
    int t = blockIdx.x * blockDim.x + threadIdx.x;
    if (t >= N * HDIM) return;
    int node = t >> 5, c = t & 31;
    float d1 = den1[node], d2 = den2[node];
    float v1 = (d1 > 0.0f ? p1[t] / d1 : 0.0f) + b1[c];
    float v2 = (d2 > 0.0f ? p2[t] / d2 : 0.0f) + b2[c];
    p1[t] = 0.5f * (v1 + v2);
}

// ---------------------------------------------------------------------------
// Final shared linear: Y = relu(Y) @ lin_W + lin_b, in place, WMMA 16x32 tiles
// ---------------------------------------------------------------------------
__global__ __launch_bounds__(256)
void final_linear_kernel(float* __restrict__ Y, const float* __restrict__ lw,
                         const float* __restrict__ lb, int N)
{
    const int lane = threadIdx.x & 31;
    const int wave = threadIdx.x >> 5;
    const int baseRow = (blockIdx.x * 8 + wave) * 16;
    if (baseRow >= N) return;

    const int r15  = lane & 15;
    const int half = lane >> 4;
    const int  rowA  = baseRow + r15;
    const bool rowOk = rowA < N;
    const float* yrow = Y + (size_t)rowA * HDIM;

    v16h a;
    if (rowOk) {
        const float4* p0 = (const float4*)(yrow + half * 16);
        const float4* p1 = (const float4*)(yrow + half * 16 + 8);
        float4 f0 = p0[0], f1 = p0[1], f2 = p1[0], f3 = p1[1];
        float t[16] = { f0.x,f0.y,f0.z,f0.w, f1.x,f1.y,f1.z,f1.w,
                        f2.x,f2.y,f2.z,f2.w, f3.x,f3.y,f3.z,f3.w };
#pragma unroll
        for (int e = 0; e < 16; ++e) a[e] = (_Float16)(t[e] > 0.0f ? t[e] : 0.0f);
    } else {
#pragma unroll
        for (int e = 0; e < 16; ++e) a[e] = (_Float16)0.0f;
    }

    v16h b0, b1;
    const float* wbase = lw + (size_t)(half * 16) * HDIM + r15;
#pragma unroll
    for (int e = 0; e < 16; ++e) {
        b0[e] = (_Float16)wbase[e * HDIM];
        b1[e] = (_Float16)wbase[e * HDIM + 16];
    }

    v8f acc0, acc1;
#pragma unroll
    for (int i = 0; i < 8; ++i) { acc0[i] = 0.0f; acc1[i] = 0.0f; }
    acc0 = __builtin_amdgcn_wmma_f32_16x16x32_f16(false, a, false, b0, (short)0, acc0, false, false);
    acc1 = __builtin_amdgcn_wmma_f32_16x16x32_f16(false, a, false, b1, (short)0, acc1, false, false);

    const float bb0 = lb[r15], bb1 = lb[16 + r15];
#pragma unroll
    for (int r = 0; r < 8; ++r) {
        int row = baseRow + half * 8 + r;
        if (row < N) {
            Y[(size_t)row * HDIM + r15]      = acc0[r] + bb0;
            Y[(size_t)row * HDIM + 16 + r15] = acc1[r] + bb1;
        }
    }
}

// ---------------------------------------------------------------------------
extern "C" void kernel_launch(void* const* d_in, const int* in_sizes, int n_in,
                              void* d_out, int out_size, void* d_ws, size_t ws_size,
                              hipStream_t stream) {
    (void)in_sizes; (void)n_in; (void)out_size; (void)ws_size;

    const float* x_author   = (const float*)d_in[0];
    const float* x_paper    = (const float*)d_in[1];
    const float* x_unit     = (const float*)d_in[2];
    const int*   writes_src = (const int*)d_in[3];
    const int*   writes_dst = (const int*)d_in[4];
    const int*   pub_src    = (const int*)d_in[5];
    const int*   pub_dst    = (const int*)d_in[6];
    const float *wr_Ws=(const float*)d_in[7],  *wr_Wd=(const float*)d_in[8],
                *wr_as=(const float*)d_in[9],  *wr_ad=(const float*)d_in[10], *wr_b=(const float*)d_in[11];
    const float *pi_Ws=(const float*)d_in[12], *pi_Wd=(const float*)d_in[13],
                *pi_as=(const float*)d_in[14], *pi_ad=(const float*)d_in[15], *pi_b=(const float*)d_in[16];
    const float *rw_Ws=(const float*)d_in[17], *rw_Wd=(const float*)d_in[18],
                *rw_as=(const float*)d_in[19], *rw_ad=(const float*)d_in[20], *rw_b=(const float*)d_in[21];
    const float *rp_Ws=(const float*)d_in[22], *rp_Wd=(const float*)d_in[23],
                *rp_as=(const float*)d_in[24], *rp_ad=(const float*)d_in[25], *rp_b=(const float*)d_in[26];
    const float *lin_W=(const float*)d_in[27], *lin_b=(const float*)d_in[28];

    // workspace layout (floats)
    float* w = (float*)d_ws;
    size_t o = 0;
    float* h_wr = w + o; o += (size_t)NA_N * HDIM;
    float* h_pi = w + o; o += (size_t)NP_N * HDIM;
    float* h_rw = w + o; o += (size_t)NP_N * HDIM;
    float* h_rp = w + o; o += (size_t)NU_N * HDIM;
    float* s_wr_s = w + o; o += NA_N;
    float* s_wr_d = w + o; o += NP_N;
    float* s_pi_s = w + o; o += NP_N;
    float* s_pi_d = w + o; o += NU_N;
    float* s_rw_s = w + o; o += NP_N;
    float* s_rw_d = w + o; o += NA_N;
    float* s_rp_s = w + o; o += NU_N;
    float* s_rp_d = w + o; o += NP_N;
    size_t zero_start = o;                    // everything below must start at 0
    unsigned* m_wr  = (unsigned*)(w + o); o += NP_N;
    float*    den_wr = w + o; o += NP_N;
    unsigned* m_pi  = (unsigned*)(w + o); o += NU_N;
    float*    den_pi = w + o; o += NU_N;
    unsigned* m_rw  = (unsigned*)(w + o); o += NA_N;
    float*    den_rw = w + o; o += NA_N;
    unsigned* m_rp  = (unsigned*)(w + o); o += NP_N;
    float*    den_rp = w + o; o += NP_N;
    float*    p2    = w + o; o += (size_t)NP_N * HDIM;

    float* out_author = (float*)d_out;
    float* out_paper  = out_author + (size_t)NA_N * HDIM;   // p1 accumulator
    float* out_unit   = out_paper  + (size_t)NP_N * HDIM;

    // zero accumulators / maxima / d_out (poisoned 0xAA by harness)
    hipMemsetAsync(d_out, 0, (size_t)(NA_N + NP_N + NU_N) * HDIM * sizeof(float), stream);
    hipMemsetAsync(w + zero_start, 0, (o - zero_start) * sizeof(float), stream);

    dim3 blk(256);
    auto gr = [](int n) { return dim3((unsigned)((n + 127) / 128)); };

    // projections + attention scores (8 fused launches; dst-side h not stored)
    proj_score_kernel<<<gr(NA_N), blk, 0, stream>>>(x_author, wr_Ws, wr_as, NA_N, 128, h_wr, s_wr_s);
    proj_score_kernel<<<gr(NP_N), blk, 0, stream>>>(x_paper,  wr_Wd, wr_ad, NP_N, 128, nullptr, s_wr_d);
    proj_score_kernel<<<gr(NP_N), blk, 0, stream>>>(x_paper,  pi_Ws, pi_as, NP_N, 128, h_pi, s_pi_s);
    proj_score_kernel<<<gr(NU_N), blk, 0, stream>>>(x_unit,   pi_Wd, pi_ad, NU_N,  64, nullptr, s_pi_d);
    proj_score_kernel<<<gr(NP_N), blk, 0, stream>>>(x_paper,  rw_Ws, rw_as, NP_N, 128, h_rw, s_rw_s);
    proj_score_kernel<<<gr(NA_N), blk, 0, stream>>>(x_author, rw_Wd, rw_ad, NA_N, 128, nullptr, s_rw_d);
    proj_score_kernel<<<gr(NU_N), blk, 0, stream>>>(x_unit,   rp_Ws, rp_as, NU_N,  64, h_rp, s_rp_s);
    proj_score_kernel<<<gr(NP_N), blk, 0, stream>>>(x_paper,  rp_Wd, rp_ad, NP_N, 128, nullptr, s_rp_d);

    const int eb = 256;
    // segment max
    edge_max_kernel<<<(EW_N + eb - 1) / eb, eb, 0, stream>>>(writes_src, writes_dst, s_wr_s, s_wr_d, m_wr, EW_N);
    edge_max_kernel<<<(EP_N + eb - 1) / eb, eb, 0, stream>>>(pub_src,    pub_dst,    s_pi_s, s_pi_d, m_pi, EP_N);
    edge_max_kernel<<<(EW_N + eb - 1) / eb, eb, 0, stream>>>(writes_dst, writes_src, s_rw_s, s_rw_d, m_rw, EW_N);
    edge_max_kernel<<<(EP_N + eb - 1) / eb, eb, 0, stream>>>(pub_dst,    pub_src,    s_rp_s, s_rp_d, m_rp, EP_N);
    // exp + weighted scatter (4 threads / edge)
    edge_accum_kernel<<<((size_t)EW_N * 4 + eb - 1) / eb, eb, 0, stream>>>(writes_src, writes_dst, s_wr_s, s_wr_d, m_wr, h_wr, den_wr, out_paper,  EW_N);
    edge_accum_kernel<<<((size_t)EP_N * 4 + eb - 1) / eb, eb, 0, stream>>>(pub_src,    pub_dst,    s_pi_s, s_pi_d, m_pi, h_pi, den_pi, out_unit,   EP_N);
    edge_accum_kernel<<<((size_t)EW_N * 4 + eb - 1) / eb, eb, 0, stream>>>(writes_dst, writes_src, s_rw_s, s_rw_d, m_rw, h_rw, den_rw, out_author, EW_N);
    edge_accum_kernel<<<((size_t)EP_N * 4 + eb - 1) / eb, eb, 0, stream>>>(pub_dst,    pub_src,    s_rp_s, s_rp_d, m_rp, h_rp, den_rp, p2,         EP_N);

    // normalize + bias
    finalize_kernel<<<((size_t)NA_N * HDIM + eb - 1) / eb, eb, 0, stream>>>(out_author, den_rw, rw_b, NA_N);
    finalize_paper_kernel<<<((size_t)NP_N * HDIM + eb - 1) / eb, eb, 0, stream>>>(out_paper, p2, den_wr, den_rp, wr_b, rp_b, NP_N);
    finalize_kernel<<<((size_t)NU_N * HDIM + eb - 1) / eb, eb, 0, stream>>>(out_unit, den_pi, pi_b, NU_N);

    // shared relu->linear over all node types, in place (WMMA)
    final_linear_kernel<<<gr(NA_N + NP_N + NU_N), blk, 0, stream>>>((float*)d_out, lin_W, lin_b, NA_N + NP_N + NU_N);
}
